// TreeLSTM_69973607186534
// MI455X (gfx1250) — matrix operand order repaired and checked by reading, not compiled, not run on hardware
//
#include <hip/hip_runtime.h>

// ---------------------------------------------------------------------------
// TreeLSTM (SPINN shift-reduce) for MI455X / gfx1250, wave32 + WMMA bf16.
//   B=64, T=256, E=300, H=512.
// Phase 1: buffer projections  x@W_x, x@W_g  -> buffers [B,T,2H] (bf16)
// Phase 2: 511-step scan; per step one [64,1024]@[1024,2560] bf16 WMMA GEMM
//          across the whole grid + elementwise gate/stack update, separated
//          by grid barriers.  W_r (bf16, 5.25MB packed) is L2-resident.
// ---------------------------------------------------------------------------

#define DEV __device__ __forceinline__

typedef __attribute__((ext_vector_type(16))) __bf16 v16bf;
typedef __attribute__((ext_vector_type(8)))  __bf16 v8bf;
typedef __attribute__((ext_vector_type(8)))  float   v8f;

static constexpr int Bb    = 64;
static constexpr int Tt    = 256;
static constexpr int Ee    = 300;
static constexpr int EP    = 320;     // E padded to multiple of 32
static constexpr int Hh    = 512;
static constexpr int TWOH  = 1024;
static constexpr int FIVEH = 2560;
static constexpr int NSTEP = 2 * Tt - 1;   // 511
static constexpr int MROWS = Bb * Tt;      // 16384

// ---- bf16 helpers ---------------------------------------------------------
DEV __bf16 f2bf(float f) {
  union { float f; unsigned u; } x; x.f = f;
  unsigned r = x.u + 0x7FFFu + ((x.u >> 16) & 1u);   // round-to-nearest-even
  union { unsigned short s; __bf16 b; } y;
  y.s = (unsigned short)(r >> 16);
  return y.b;
}
DEV float bf2f(__bf16 b) {
  union { unsigned short s; __bf16 b; } y; y.b = b;
  union { float f; unsigned u; } x; x.u = ((unsigned)y.s) << 16;
  return x.f;
}
DEV float sigm(float z) { return 1.0f / (1.0f + __expf(-z)); }

// ---- monotonic grid barrier (counter reset by init kernel each launch) ----
DEV void gbar(int* bar, int gen, int nblocks) {
  __threadfence();
  __syncthreads();
  if (threadIdx.x == 0) {
    __hip_atomic_fetch_add(bar, 1, __ATOMIC_RELEASE, __HIP_MEMORY_SCOPE_AGENT);
    while (__hip_atomic_load(bar, __ATOMIC_ACQUIRE, __HIP_MEMORY_SCOPE_AGENT) <
           gen * nblocks) {
      __builtin_amdgcn_s_sleep(2);
    }
  }
  __syncthreads();
  __threadfence();
}

// ---------------------------------------------------------------------------
// Init control block: [0]=barrier, +64 sp0, +128 sp1, +192 bp0, +256 bp1
// ---------------------------------------------------------------------------
__global__ void init_kernel(int* ctrl) {
  int tid = threadIdx.x;
  if (tid == 0) ctrl[0] = 0;
  if (tid < Bb) {
    ctrl[64 + tid]  = 0;    // sp0
    ctrl[128 + tid] = 0;    // sp1
    ctrl[192 + tid] = Tt;   // bp0
    ctrl[256 + tid] = Tt;   // bp1
  }
}

// ---------------------------------------------------------------------------
// Pack x [16384,300] f32 -> [16384,320] bf16 (zero-padded K)
// ---------------------------------------------------------------------------
__global__ void pack_x_kernel(const float* __restrict__ x, __bf16* __restrict__ xpad) {
  int idx = blockIdx.x * blockDim.x + threadIdx.x;
  if (idx >= MROWS * EP) return;
  int m = idx / EP, k = idx % EP;
  float v = (k < Ee) ? x[(size_t)m * Ee + k] : 0.0f;
  xpad[idx] = f2bf(v);
}

// ---------------------------------------------------------------------------
// Pack W_x|W_g (fused N=1024) into WMMA B-operand layout:
//   id = ((kt*64 + nt)*32 + lane)*16 + h ;  K = kt*32 + (lane<16?0:16) + h
//   n = nt*16 + (lane&15); n<512 -> W_x col n, else W_g col n-512
// ---------------------------------------------------------------------------
__global__ void pack_wxg_kernel(const float* __restrict__ W_x,
                                const float* __restrict__ W_g,
                                __bf16* __restrict__ wxg) {
  int id = blockIdx.x * blockDim.x + threadIdx.x;
  if (id >= 10 * 64 * 32 * 16) return;
  int h    = id & 15;
  int lane = (id >> 4) & 31;
  int ktnt = id >> 9;
  int nt   = ktnt & 63;
  int kt   = ktnt >> 6;
  int K = kt * 32 + ((lane < 16) ? 0 : 16) + h;
  int n = nt * 16 + (lane & 15);
  float v = 0.0f;
  if (K < Ee) v = (n < Hh) ? W_x[(size_t)K * Hh + n]
                           : W_g[(size_t)K * Hh + (n - Hh)];
  wxg[id] = f2bf(v);
}

// ---------------------------------------------------------------------------
// Pack W_r [1024,2560] into WMMA B-operand layout:
//   id = ((kt*160 + nt)*32 + lane)*16 + h
// ---------------------------------------------------------------------------
__global__ void pack_wr_kernel(const float* __restrict__ W_r, __bf16* __restrict__ wrp) {
  int id = blockIdx.x * blockDim.x + threadIdx.x;
  if (id >= 32 * 160 * 32 * 16) return;
  int h    = id & 15;
  int lane = (id >> 4) & 31;
  int ktnt = id >> 9;
  int nt   = ktnt % 160;
  int kt   = ktnt / 160;
  int K = kt * 32 + ((lane < 16) ? 0 : 16) + h;
  int n = nt * 16 + (lane & 15);
  wrp[id] = f2bf(W_r[(size_t)K * FIVEH + n]);
}

// ---------------------------------------------------------------------------
// Phase 1: buffer projections.  Each wave owns one 16-row M tile and a pair
// of N tiles (cx tile nt, gx tile nt+32) so the epilogue can fuse the gate.
//   c_buf = x@W_x + b_x ; h_buf = sigmoid(x@W_g + b_g) * tanh(c_buf)
// buffers layout [B*T, 2H] bf16: [0:H)=h, [H:2H)=c
// ---------------------------------------------------------------------------
__global__ void buffer_proj_kernel(const __bf16* __restrict__ xpad,
                                   const __bf16* __restrict__ wxg,
                                   const float* __restrict__ b_x,
                                   const float* __restrict__ b_g,
                                   __bf16* __restrict__ buffers) {
  const int lane = threadIdx.x & 31;
  const int wave = (blockIdx.x * blockDim.x + threadIdx.x) >> 5;  // 0..32767
  const int mt   = wave >> 5;                                     // 0..1023
  const int ntp  = wave & 31;                                     // 0..31
  const int mrow  = mt * 16 + (lane & 15);
  const int kbase = (lane < 16) ? 0 : 16;
  const __bf16* arow = xpad + (size_t)mrow * EP;

  v8f c0 = {0.f,0.f,0.f,0.f,0.f,0.f,0.f,0.f};
  v8f c1 = {0.f,0.f,0.f,0.f,0.f,0.f,0.f,0.f};

  for (int kt = 0; kt < 10; ++kt) {
    const int k0 = kt * 32;
    v8bf lo = *(const v8bf*)(arow + k0 + kbase);
    v8bf hi = *(const v8bf*)(arow + k0 + 8 + kbase);
    v16bf a;
#pragma unroll
    for (int j = 0; j < 8; ++j) { a[j] = lo[j]; a[8 + j] = hi[j]; }
    v16bf bm0 = *(const v16bf*)(wxg + ((size_t)(kt * 64 + ntp)      * 32 + lane) * 16);
    v16bf bm1 = *(const v16bf*)(wxg + ((size_t)(kt * 64 + ntp + 32) * 32 + lane) * 16);
    c0 = __builtin_amdgcn_wmma_f32_16x16x32_bf16(false, a, false, bm0, (short)0, c0, false, false);
    c1 = __builtin_amdgcn_wmma_f32_16x16x32_bf16(false, a, false, bm1, (short)0, c1, false, false);
  }

  const int col = ntp * 16 + (lane & 15);
  const float bx = b_x[col], bg = b_g[col];
#pragma unroll
  for (int r = 0; r < 8; ++r) {
    int m = mt * 16 + r + ((lane < 16) ? 0 : 8);
    float cbuf = c0[r] + bx;
    float hbuf = sigm(c1[r] + bg) * tanhf(cbuf);
    buffers[(size_t)m * TWOH + col]      = f2bf(hbuf);
    buffers[(size_t)m * TWOH + Hh + col] = f2bf(cbuf);
  }
}

// ---------------------------------------------------------------------------
// Phase 2: persistent scan kernel.  64 blocks x 256 threads (512 wave32s).
// Per step:
//   A) proj[64,2560] = concat(h_l,h_r) @ W_r      (640 WMMA tiles, K=1024)
//   B) gates + stack update (bf16 stack), ping-pong sp/bp
// ---------------------------------------------------------------------------
__global__ void scan_kernel(__bf16* __restrict__ stack,
                            const __bf16* __restrict__ buffers,
                            const __bf16* __restrict__ wrp,
                            float* __restrict__ proj,
                            const int* __restrict__ trans,
                            const float* __restrict__ b_r,
                            int* __restrict__ ctrl) {
  int* bar = ctrl;
  int* sp0 = ctrl + 64;
  int* sp1 = ctrl + 128;
  int* bp0 = ctrl + 192;
  int* bp1 = ctrl + 256;

  const int lane = threadIdx.x & 31;
  const int wid  = (blockIdx.x * blockDim.x + threadIdx.x) >> 5;   // 0..511
  const int gtid = blockIdx.x * blockDim.x + threadIdx.x;          // 0..16383
  int nbar = 0;

  for (int t = 0; t < NSTEP; ++t) {
    int* spc = (t & 1) ? sp1 : sp0;
    int* bpc = (t & 1) ? bp1 : bp0;
    int* spn = (t & 1) ? sp0 : sp1;
    int* bpn = (t & 1) ? bp0 : bp1;

    // ---- Phase A: GEMM -> proj -------------------------------------------
    for (int tile = wid; tile < 4 * 160; tile += 512) {
      const int mt = tile / 160;
      const int nt = tile % 160;
      const int b  = mt * 16 + (lane & 15);       // batch index = M row
      const int sp = spc[b];
      const int rl = (sp - 2 < 0) ? 0 : sp - 2;
      const int rr = (sp - 1 < 0) ? 0 : sp - 1;
      const __bf16* rowl = stack + ((size_t)b * Tt + rl) * TWOH;   // h at [0:512)
      const __bf16* rowr = stack + ((size_t)b * Tt + rr) * TWOH;
      const int kbase = (lane < 16) ? 0 : 16;

      v8f acc = {0.f,0.f,0.f,0.f,0.f,0.f,0.f,0.f};
      for (int kt = 0; kt < 32; ++kt) {
        const int ks0 = kt * 32 + kbase;
        const int ks1 = ks0 + 8;
        const __bf16* p0 = (ks0 < Hh) ? (rowl + ks0) : (rowr + (ks0 - Hh));
        const __bf16* p1 = (ks1 < Hh) ? (rowl + ks1) : (rowr + (ks1 - Hh));
        v8bf lo = *(const v8bf*)p0;
        v8bf hi = *(const v8bf*)p1;
        v16bf a;
#pragma unroll
        for (int j = 0; j < 8; ++j) { a[j] = lo[j]; a[8 + j] = hi[j]; }
        const __bf16* wptr = wrp + ((size_t)(kt * 160 + nt) * 32 + lane) * 16;
        v16bf bm = *(const v16bf*)wptr;
        if (kt + 1 < 32) {
          __builtin_prefetch(wrp + ((size_t)((kt + 1) * 160 + nt) * 32 + lane) * 16, 0, 1);
        }
        acc = __builtin_amdgcn_wmma_f32_16x16x32_bf16(false, a, false, bm, (short)0, acc, false, false);
      }
#pragma unroll
      for (int r = 0; r < 8; ++r) {
        int mm = mt * 16 + r + ((lane < 16) ? 0 : 8);
        int n  = nt * 16 + (lane & 15);
        proj[(size_t)mm * FIVEH + n] = acc[r];
      }
    }
    gbar(bar, ++nbar, 64);

    // ---- Phase B: gates + stack update -----------------------------------
    for (int item = gtid; item < Bb * Hh; item += 16384) {
      const int b = item >> 9;
      const int j = item & 511;
      const int tr = trans[t * Bb + b];
      const int sp = spc[b];
      const int bp = bpc[b];
      float h, c;
      if (tr == 0) {  // shift from buffer top
        int row = (bp - 1 < 0) ? 0 : bp - 1;
        const __bf16* src = buffers + ((size_t)b * Tt + row) * TWOH;
        h = bf2f(src[j]);
        c = bf2f(src[Hh + j]);
      } else {        // reduce via N-ary cell
        int rl = (sp - 2 < 0) ? 0 : sp - 2;
        int rr = (sp - 1 < 0) ? 0 : sp - 1;
        const float* pr = proj + (size_t)b * FIVEH;
        float gi  = sigm (pr[j]        + b_r[j]);
        float gfl = sigm (pr[512 + j]  + b_r[512 + j]);
        float gfr = sigm (pr[1024 + j] + b_r[1024 + j]);
        float gg  = tanhf(pr[1536 + j] + b_r[1536 + j]);
        float go  = sigm (pr[2048 + j] + b_r[2048 + j]);
        float cl = bf2f(stack[((size_t)b * Tt + rl) * TWOH + Hh + j]);
        float cr = bf2f(stack[((size_t)b * Tt + rr) * TWOH + Hh + j]);
        c = gi * gg + gfl * cl + gfr * cr;
        h = go * tanhf(c);
      }
      const int nsp = (tr == 0) ? sp + 1 : sp - 1;
      const int w   = (nsp - 1 < 0) ? 0 : nsp - 1;
      __bf16* dst = stack + ((size_t)b * Tt + w) * TWOH;
      dst[j]      = f2bf(h);
      dst[Hh + j] = f2bf(c);
      if (j == 0) {
        spn[b] = nsp;
        bpn[b] = (tr == 0) ? bp - 1 : bp;
      }
    }
    gbar(bar, ++nbar, 64);
  }
}

// ---------------------------------------------------------------------------
// Output: root h states [B, H] fp32 (final sp lives in sp1 after 511 steps)
// ---------------------------------------------------------------------------
__global__ void output_kernel(const __bf16* __restrict__ stack,
                              const int* __restrict__ ctrl,
                              float* __restrict__ out) {
  int idx = blockIdx.x * blockDim.x + threadIdx.x;
  if (idx >= Bb * Hh) return;
  int b = idx >> 9, j = idx & 511;
  int sp = ctrl[128 + b];                    // sp1
  int row = (sp - 1 < 0) ? 0 : sp - 1;
  out[idx] = bf2f(stack[((size_t)b * Tt + row) * TWOH + j]);
}

// ---------------------------------------------------------------------------
// Launch
// ---------------------------------------------------------------------------
extern "C" void kernel_launch(void* const* d_in, const int* in_sizes, int n_in,
                              void* d_out, int out_size, void* d_ws, size_t ws_size,
                              hipStream_t stream) {
  const float* x     = (const float*)d_in[0];
  const int*   trans = (const int*)  d_in[1];
  const float* W_x   = (const float*)d_in[2];
  const float* b_x   = (const float*)d_in[3];
  const float* W_g   = (const float*)d_in[4];
  const float* b_g   = (const float*)d_in[5];
  const float* W_r   = (const float*)d_in[6];
  const float* b_r   = (const float*)d_in[7];
  float* out = (float*)d_out;

  char* ws = (char*)d_ws;
  const size_t SZ_XPAD = (size_t)MROWS * EP * 2;           // 10,485,760
  const size_t SZ_WXG  = (size_t)10 * 64 * 32 * 16 * 2;    //    655,360
  const size_t SZ_WR   = (size_t)32 * 160 * 32 * 16 * 2;   //  5,242,880
  const size_t SZ_BUF  = (size_t)MROWS * TWOH * 2;         // 33,554,432
  const size_t SZ_STK  = (size_t)MROWS * TWOH * 2;         // 33,554,432
  const size_t SZ_PROJ = (size_t)Bb * FIVEH * 4;           //    655,360

  __bf16* xpad    = (__bf16*)(ws);
  __bf16* wxg     = (__bf16*)(ws + SZ_XPAD);
  __bf16* wrp     = (__bf16*)(ws + SZ_XPAD + SZ_WXG);
  __bf16* buffers = (__bf16*)(ws + SZ_XPAD + SZ_WXG + SZ_WR);
  __bf16* stack   = (__bf16*)(ws + SZ_XPAD + SZ_WXG + SZ_WR + SZ_BUF);
  float*  proj    = (float*) (ws + SZ_XPAD + SZ_WXG + SZ_WR + SZ_BUF + SZ_STK);
  int*    ctrl    = (int*)   (ws + SZ_XPAD + SZ_WXG + SZ_WR + SZ_BUF + SZ_STK + SZ_PROJ);

  init_kernel<<<1, 256, 0, stream>>>(ctrl);
  pack_x_kernel<<<(MROWS * EP + 255) / 256, 256, 0, stream>>>(x, xpad);
  pack_wxg_kernel<<<(10 * 64 * 32 * 16 + 255) / 256, 256, 0, stream>>>(W_x, W_g, wxg);
  pack_wr_kernel<<<(32 * 160 * 32 * 16 + 255) / 256, 256, 0, stream>>>(W_r, wrp);
  buffer_proj_kernel<<<4096, 256, 0, stream>>>(xpad, wxg, b_x, b_g, buffers);
  scan_kernel<<<64, 256, 0, stream>>>(stack, buffers, wrp, proj, trans, b_r, ctrl);
  output_kernel<<<(Bb * Hh + 255) / 256, 256, 0, stream>>>(stack, ctrl, out);

  (void)in_sizes; (void)n_in; (void)out_size; (void)ws_size;
}
